// OmostDenseDiffusionCrossAttention_14809047236989
// MI455X (gfx1250) — matrix operand order repaired and checked by reading, not compiled, not run on hardware
//
#include <hip/hip_runtime.h>
#include <stdint.h>

#define BB   4
#define SQn  4096
#define SKn  462
#define HHn  10
#define DHn  64
#define HD   (HHn * DHn)
#define KCH  32
#define NCH  15                  // ceil(462/32)
#define SKP  (NCH * KCH)         // 480 keys, zero-padded
#define TAIL ((NCH - 1) * KCH)   // 448: base of last (partial) chunk
#define KS   72                  // klds row stride (bf16): 144B rows, 16B aligned
#define VS   488                 // vtlds row stride: 976B rows, full 64-bank cycle
#define PS   40                  // plds row stride: 80B rows
#define WPB  8                   // waves (q-tiles) per block

typedef __attribute__((ext_vector_type(16))) __bf16 v16bf;
typedef __attribute__((ext_vector_type(8)))  float  v8f;

static __device__ __forceinline__ unsigned short f2bf(float x) {
  return __builtin_bit_cast(unsigned short, (__bf16)x);   // native cvt, RNE
}

// DPP16 row_ror butterflies within each 16-lane row (wave32 => two rows)
template <int CTRL>
static __device__ __forceinline__ float dppf(float x) {
  return __int_as_float(__builtin_amdgcn_update_dpp(
      0, __float_as_int(x), CTRL, 0xF, 0xF, true));
}
static __device__ __forceinline__ float redmax16(float x) {
  x = fmaxf(x, dppf<0x128>(x));   // row_ror:8
  x = fmaxf(x, dppf<0x124>(x));   // row_ror:4
  x = fmaxf(x, dppf<0x122>(x));   // row_ror:2
  x = fmaxf(x, dppf<0x121>(x));   // row_ror:1
  return x;
}
static __device__ __forceinline__ float redsum16(float x) {
  x += dppf<0x128>(x);
  x += dppf<0x124>(x);
  x += dppf<0x122>(x);
  x += dppf<0x121>(x);
  return x;
}

union Frag { v16bf vv; uint4 q[2]; unsigned short u[16]; };

__global__ __launch_bounds__(WPB * 32, 4)
void omost_xattn_kernel(const float* __restrict__ q,
                        const float* __restrict__ k,
                        const float* __restrict__ v,
                        const unsigned char* __restrict__ mask_bool,
                        const float* __restrict__ mask_scale,
                        float* __restrict__ out)
{
  __shared__ __align__(16) unsigned short klds[SKP * KS];      // K   [key][dh]  bf16
  __shared__ __align__(16) unsigned short vtlds[DHn * VS];     // V^T [dh][key]  bf16
  __shared__ __align__(16) unsigned short plds[WPB * 16 * PS]; // P tiles, per wave

  const int tid  = threadIdx.x;
  const int w    = tid >> 5;
  const int lane = tid & 31;
  const int half = lane >> 4;
  const int lq   = lane & 15;

  const int h = blockIdx.y;
  const int b = blockIdx.z;

  const float scale = 0.125f;     // 1/sqrt(64), folded into Q
  const float NEG   = -1.0e30f;   // rowmax init and the compile-time dead tile

  // ---- cooperative K/V staging: f32 global -> bf16 LDS (V transposed) ----
  for (int idx = tid; idx < SKP * DHn; idx += WPB * 32) {
    int key = idx >> 6;
    int d   = idx & 63;
    float kv = 0.f, vv = 0.f;
    if (key < SKn) {
      size_t g = (size_t)(b * SKn + key) * HD + h * DHn + d;
      kv = k[g];
      vv = v[g];
    }
    klds[key * KS + d]  = f2bf(kv);
    vtlds[d * VS + key] = f2bf(vv);
  }
  __syncthreads();

  // ---- per-wave flash attention over one 16-query tile ----
  const int qbase = (blockIdx.x * WPB + w) * 16;

  // Q A-fragments (16x32 bf16, ISA 16-bit A layout), pre-scaled by 1/sqrt(dh)
  const float* qrow = q + (size_t)(b * SQn + qbase + lq) * HD + h * DHn;
  v16bf aq[2];
#pragma unroll
  for (int c = 0; c < 2; ++c) {
    Frag t;
#pragma unroll
    for (int i = 0; i < 8; ++i) {
      int kd = c * 32 + half * 8 + ((i < 4) ? 2 * i : 16 + 2 * (i - 4));
      float2 f = *(const float2*)(qrow + kd);
      t.u[2 * i]     = f2bf(f.x * scale);
      t.u[2 * i + 1] = f2bf(f.y * scale);
    }
    aq[c] = t.vv;
  }

  v8f o[4] = {};
  float rowmax[8], rowsum[8];
#pragma unroll
  for (int r = 0; r < 8; ++r) { rowmax[r] = NEG; rowsum[r] = 0.f; }

  // Per-lane base pointers; in-chunk offsets r*SK + t*16 are immediates.
  const int rowb = ((b * HHn + h) * SQn + qbase + half * 8) * SKn;
  const float*         msp = mask_scale + rowb + lq;
  const unsigned char* mbp = mask_bool  + rowb + lq;
  // Tail chunk: column clamped in-bounds; invalid lanes zeroed multiplicatively.
  const int tcol = (TAIL + lq < SKn) ? (TAIL + lq) : (SKn - 1);
  const float*         mspT = mask_scale + rowb + tcol;
  const unsigned char* mbpT = mask_bool  + rowb + tcol;
  const float tailValid = (TAIL + lq < SKn) ? 1.f : 0.f;

  unsigned short* pw = &plds[w * 16 * PS];

  // One attention chunk of 32 keys. `tailf` is a constant after inlining.
  auto chunk = [&](const int kc, const bool tailf) __attribute__((always_inline)) {
    if (!tailf) {   // speculative prefetch of next chunk's mask rows
#pragma unroll
      for (int r = 0; r < 8; ++r) {
        __builtin_prefetch(msp + r * SKn + kc + KCH, 0, 1);
        __builtin_prefetch(mbp + r * SKn + kc + KCH, 0, 1);
      }
    }

    float sv[2][8];   // score * mask_scale (all entries)
    float fm[2][8];   // multiplicative mask factor (0.0 / 1.0)
#pragma unroll
    for (int t = 0; t < 2; ++t) {
      if (tailf && t == 1) {              // fully OOB tile: compile-time dead
#pragma unroll
        for (int r = 0; r < 8; ++r) { sv[1][r] = NEG; fm[1][r] = 0.f; }
      } else {
        const int kt = kc + t * 16;
        v8f s = {};
#pragma unroll
        for (int c = 0; c < 2; ++c) {
          Frag bt;
          const unsigned short* kr = &klds[(kt + lq) * KS + c * 32 + half * 8];
          bt.q[0] = *(const uint4*)kr;
          bt.q[1] = *(const uint4*)(kr + 16);
          s = __builtin_amdgcn_wmma_f32_16x16x32_bf16(
                  false, aq[c], false, bt.vv, (short)0, s, false, false);
        }
#pragma unroll
        for (int r = 0; r < 8; ++r) {     // branchless mask application
          float ms, mb;
          if (tailf) {
            ms = mspT[r * SKn];
            mb = (float)mbpT[r * SKn] * tailValid;
          } else {
            ms = msp[r * SKn + kt];
            mb = (float)mbp[r * SKn + kt];
          }
          sv[t][r] = s[r] * ms;           // unconditional use: no load sinking
          fm[t][r] = mb;
        }
      }
    }

    // online softmax (max over ALL entries is safe; mask applied on exp)
#pragma unroll
    for (int r = 0; r < 8; ++r) {
      float cm = redmax16(fmaxf(sv[0][r], sv[1][r]));
      float nm = fmaxf(rowmax[r], cm);
      float corr = __expf(rowmax[r] - nm);
      rowmax[r] = nm;
      float p0 = __expf(sv[0][r] - nm) * fm[0][r];
      float p1 = __expf(sv[1][r] - nm) * fm[1][r];
      rowsum[r] = rowsum[r] * corr + redsum16(p0 + p1);
#pragma unroll
      for (int nb = 0; nb < 4; ++nb) o[nb][r] *= corr;
      int m = half * 8 + r;
      pw[m * PS + lq]      = f2bf(p0);
      pw[m * PS + 16 + lq] = f2bf(p1);
    }

    // A-fragment of P (C-layout -> A-layout via per-wave LDS tile)
    Frag ap;
    ap.q[0] = *(const uint4*)&pw[lq * PS + half * 8];
    ap.q[1] = *(const uint4*)&pw[lq * PS + half * 8 + 16];

    // O += P(16x32) x V(32x64); V^T fragments from LDS (b128 x2)
#pragma unroll
    for (int nb = 0; nb < 4; ++nb) {
      Frag bv;
      const unsigned short* vr = &vtlds[(nb * 16 + lq) * VS + kc + half * 8];
      bv.q[0] = *(const uint4*)vr;
      bv.q[1] = *(const uint4*)(vr + 16);
      o[nb] = __builtin_amdgcn_wmma_f32_16x16x32_bf16(
                  false, ap.vv, false, bv.vv, (short)0, o[nb], false, false);
    }
  };

  // 14 full chunks with a bounded pipelining window, then the peeled tail.
#pragma unroll 2
  for (int ic = 0; ic < NCH - 1; ++ic)
    chunk(ic * KCH, false);
  chunk(TAIL, true);

  // ---- normalize and store (D layout: lane = column, VGPR slot = row) ----
#pragma unroll
  for (int r = 0; r < 8; ++r) {
    float inv = rowsum[r] > 0.f ? 1.0f / rowsum[r] : 0.f;
    int m = half * 8 + r;
    float* orow = out + (size_t)(b * SQn + qbase + m) * HD + h * DHn;
#pragma unroll
    for (int nb = 0; nb < 4; ++nb)
      orow[nb * 16 + lq] = o[nb][r] * inv;
  }
}

extern "C" void kernel_launch(void* const* d_in, const int* in_sizes, int n_in,
                              void* d_out, int out_size, void* d_ws, size_t ws_size,
                              hipStream_t stream) {
  const float*         q          = (const float*)d_in[0];
  const float*         k          = (const float*)d_in[1];
  const float*         v          = (const float*)d_in[2];
  const unsigned char* mask_bool  = (const unsigned char*)d_in[3];
  const float*         mask_scale = (const float*)d_in[4];
  // d_in[5] = heads (compile-time constant HHn)
  float* out = (float*)d_out;

  dim3 grid(SQn / 16 / WPB, HHn, BB);   // 32 x 10 x 4 blocks, 8 waves each
  omost_xattn_kernel<<<grid, WPB * 32, 0, stream>>>(q, k, v, mask_bool, mask_scale, out);
}
